// GaussianUpsampling_11381663334992
// MI455X (gfx1250) — matrix operand order repaired and checked by reading, not compiled, not run on hardware
//
#include <hip/hip_runtime.h>
#include <hip/hip_bf16.h>

typedef __attribute__((ext_vector_type(16))) __bf16 v16bf;
typedef __attribute__((ext_vector_type(8)))  __bf16 v8bf;
typedef __attribute__((ext_vector_type(2)))  __bf16 v2bf;
typedef __attribute__((ext_vector_type(8)))  float  v8f;
typedef __attribute__((ext_vector_type(2)))  float  v2f;

#define HMAX 656          // per-h LDS staging capacity (H=600 in harness)
#define BM   128
#define BN   256
#define BK   32
#define KS   40           // LDS K-stride (32 + 8 pad), 80B rows -> 16B aligned
#define NSUB 8            // 16x16 subtiles per wave along N (BN/2/16)
#define NTHREADS 512      // 16 wave32 waves: 8 (M) x 2 (N)
#define A_ITERS ((BM * (BK / 2)) / NTHREADS)   // 4, exact
#define B_ITERS (((BK / 2) * BN) / NTHREADS)   // 8, exact

// ---------------- Kernel 1: per-phoneme precompute ----------------
__global__ void gu_precompute(const float* __restrict__ dur,
                              const float* __restrict__ vars,
                              float* __restrict__ c_out,
                              float* __restrict__ coef_out,
                              float* __restrict__ base_out,
                              int H) {
    __shared__ float sd[HMAX];
    __shared__ float sc[HMAX];
    const int b = blockIdx.x;
    const int tid = threadIdx.x;
    for (int h = tid; h < H; h += blockDim.x) sd[h] = dur[(long)b * H + h];
    __syncthreads();
    if (tid == 0) {
        float acc = 0.f;
        for (int h = 0; h < H; ++h) { acc += sd[h]; sc[h] = acc; }
    }
    __syncthreads();
    const float LOG2PI = 1.8378770664093453f;
    for (int h = tid; h < H; h += blockDim.x) {
        float v = vars[(long)b * H + h];
        c_out[(long)b * H + h]    = sc[h] - 0.5f * sd[h];
        coef_out[(long)b * H + h] = -0.5f / v;
        base_out[(long)b * H + h] = -0.5f * (LOG2PI + __logf(v));
    }
}

// ---------------- Kernel 2: per-(b,t) softmax stats ----------------
__global__ void gu_stats(const float* __restrict__ c_in,
                         const float* __restrict__ coef_in,
                         const float* __restrict__ base_in,
                         const int* __restrict__ lengths,
                         float* __restrict__ mx_out,
                         float* __restrict__ sinv_out,
                         int H, int T) {
    __shared__ float shc[HMAX], shcf[HMAX], shbs[HMAX];
    const int b = blockIdx.y;
    const int tid = threadIdx.x;
    for (int h = tid; h < H; h += blockDim.x) {
        shc[h]  = c_in[(long)b * H + h];
        shcf[h] = coef_in[(long)b * H + h];
        shbs[h] = base_in[(long)b * H + h];
    }
    __syncthreads();
    int t = blockIdx.x * blockDim.x + tid;
    if (t >= T) return;
    int len = lengths[b]; if (len > H) len = H; if (len < 1) len = 1;
    const float tf = (float)t;
    float mx = -3.4e38f;
    for (int h = 0; h < len; ++h) {
        float diff = tf - shc[h];
        float l = shbs[h] + shcf[h] * diff * diff;
        mx = fmaxf(mx, l);
    }
    float s = 0.f;
    for (int h = 0; h < len; ++h) {
        float diff = tf - shc[h];
        float l = shbs[h] + shcf[h] * diff * diff;
        s += __expf(l - mx);
    }
    mx_out[(long)b * T + t] = mx;
    sinv_out[(long)b * T + t] = 1.0f / s;
}

// ---------------- Kernel 3: fused softmax-weight x encoder matmul (bf16 WMMA) ----------------
union V16U { v16bf v; v8bf h[2]; };
union V2U  { v2bf v; __bf16 b[2]; };

__global__ __launch_bounds__(NTHREADS)
void gu_matmul(const float* __restrict__ enc,
               const int* __restrict__ lengths,
               const float* __restrict__ c_in,
               const float* __restrict__ coef_in,
               const float* __restrict__ base_in,
               const float* __restrict__ mx_in,
               const float* __restrict__ sinv_in,
               float* __restrict__ out,
               int H, int D, int T) {
    __shared__ __align__(16) __bf16 ldsA[BM * KS];   // P tile  [t_local][k]
    __shared__ __align__(16) __bf16 ldsB[BN * KS];   // E tile transposed [n][k]
    __shared__ __align__(8) float s_c[HMAX], s_cf[HMAX], s_bs[HMAX];
    __shared__ float s_m[BM], s_is[BM];

    const int tid = threadIdx.x;
    const int d0 = blockIdx.x * BN;
    const int t0 = blockIdx.y * BM;
    const int b  = blockIdx.z;

    for (int h = tid; h < H; h += NTHREADS) {
        s_c[h]  = c_in[(long)b * H + h];
        s_cf[h] = coef_in[(long)b * H + h];
        s_bs[h] = base_in[(long)b * H + h];
    }
    if (tid < BM) {
        int t = t0 + tid;
        s_m[tid]  = (t < T) ? mx_in[(long)b * T + t] : 0.f;
        s_is[tid] = (t < T) ? sinv_in[(long)b * T + t] : 0.f;
    }
    int len = lengths[b]; if (len > H) len = H;
    __syncthreads();

    const int wave   = tid >> 5;
    const int lane   = tid & 31;
    const int wm     = wave & 7;       // 0..7 -> 16-row slice of BM=128
    const int wn     = wave >> 3;      // 0..1 -> 128-col slice of BN=256
    const int lane16 = lane & 15;
    const int hi     = lane >> 4;      // 0 or 1 (lane half)

    v8f acc[NSUB] = {};

    const long encBase = (long)b * H * (long)D;

    for (int k0 = 0; k0 < H; k0 += BK) {
        __syncthreads();
        // ---- Fill A: softmax weights, branchless, packed bf16 pairs ----
        // 2048 pairs over 512 threads: exactly 4 iterations, no guards, EXEC stays full.
#pragma unroll
        for (int i = 0; i < A_ITERS; ++i) {
            int idx = tid + i * NTHREADS;
            int tl = idx >> 4;
            int kp = idx & 15;
            int h  = k0 + 2 * kp;          // even, 8B-aligned for float2 loads
            int t  = t0 + tl;
            v2f cc = *(const v2f*)(s_c + h);
            v2f cf = *(const v2f*)(s_cf + h);
            v2f bs = *(const v2f*)(s_bs + h);
            float tf = (float)t;
            float m  = s_m[tl];
            float is = s_is[tl];
            float d0f = tf - cc.x, d1f = tf - cc.y;
            float p0 = __expf(bs.x + cf.x * d0f * d0f - m) * is;
            float p1 = __expf(bs.y + cf.y * d1f * d1f - m) * is;
            bool tok = (t < T);
            p0 = (tok && (h     < len)) ? p0 : 0.f;
            p1 = (tok && (h + 1 < len)) ? p1 : 0.f;
            V2U pk;
            pk.b[0] = (__bf16)p0;
            pk.b[1] = (__bf16)p1;
            *(v2bf*)(ldsA + tl * KS + 2 * kp) = pk.v;
        }
        // ---- Fill B: two coalesced global rows per thread, packed bf16 pair ----
        // 4096 pairs over 512 threads: exactly 8 iterations, no guards.
#pragma unroll
        for (int i = 0; i < B_ITERS; ++i) {
            int idx = tid + i * NTHREADS;
            int n  = idx & (BN - 1);
            int kp = idx >> 8;             // BN == 256
            int h  = k0 + 2 * kp;
            int d  = d0 + n;
            int hc0 = (h     < H) ? h     : (H - 1);
            int hc1 = (h + 1 < H) ? h + 1 : (H - 1);
            int dc  = (d < D) ? d : (D - 1);
            float e0 = enc[encBase + (long)hc0 * D + dc];
            float e1 = enc[encBase + (long)hc1 * D + dc];
            e0 = ((h     < H) && (d < D)) ? e0 : 0.f;
            e1 = ((h + 1 < H) && (d < D)) ? e1 : 0.f;
            V2U pk;
            pk.b[0] = (__bf16)e0;
            pk.b[1] = (__bf16)e1;
            *(v2bf*)(ldsB + n * KS + 2 * kp) = pk.v;
        }
        __syncthreads();

        // ---- A fragment: lane m = lane16; K runs {0..15} or {16..31} ----
        const int mrow = wm * 16 + lane16;
        const int kbA  = hi ? 16 : 0;
        V16U af;
        af.h[0] = *(const v8bf*)(ldsA + mrow * KS + kbA);
        af.h[1] = *(const v8bf*)(ldsA + mrow * KS + kbA + 8);

        // ---- B fragments + WMMA burst (reuse_a across identical instructions) ----
        const int kbB = hi ? 16 : 0;
        {
            V16U bfr;
            int n = wn * 128 + lane16;
            bfr.h[0] = *(const v8bf*)(ldsB + n * KS + kbB);
            bfr.h[1] = *(const v8bf*)(ldsB + n * KS + kbB + 8);
            acc[0] = __builtin_amdgcn_wmma_f32_16x16x32_bf16(
                false, af.v, false, bfr.v, (short)0, acc[0], false, false);
        }
#pragma unroll
        for (int s = 1; s < NSUB; ++s) {
            V16U bfr;
            int n = wn * 128 + s * 16 + lane16;
            bfr.h[0] = *(const v8bf*)(ldsB + n * KS + kbB);
            bfr.h[1] = *(const v8bf*)(ldsB + n * KS + kbB + 8);
            acc[s] = __builtin_amdgcn_wmma_f32_16x16x32_bf16(
                false, af.v, false, bfr.v, (short)0, acc[s], true, false);
        }
    }

    // ---- Write D tile: VGPR v -> M = v + 8*hi, lane16 -> N ----
    const int mbase = t0 + wm * 16 + (hi ? 8 : 0);
#pragma unroll
    for (int s = 0; s < NSUB; ++s) {
        int d = d0 + wn * 128 + s * 16 + lane16;
        if (d < D) {
#pragma unroll
            for (int v = 0; v < 8; ++v) {
                int t = mbase + v;
                if (t < T) out[((long)b * T + t) * (long)D + d] = acc[s][v];
            }
        }
    }
}

extern "C" void kernel_launch(void* const* d_in, const int* in_sizes, int n_in,
                              void* d_out, int out_size, void* d_ws, size_t ws_size,
                              hipStream_t stream) {
    const float* enc     = (const float*)d_in[0];
    const float* dur     = (const float*)d_in[1];
    const float* vars    = (const float*)d_in[2];
    const int*   lengths = (const int*)d_in[3];
    float* out = (float*)d_out;

    const int B = in_sizes[3];
    const int H = in_sizes[1] / B;
    const int D = in_sizes[0] / in_sizes[1];
    const int T = out_size / (B * D);

    // Workspace layout (floats): c[B*H] | coef[B*H] | base[B*H] | mx[B*T] | sinv[B*T]
    float* ws    = (float*)d_ws;
    float* c_p   = ws;
    float* coef  = c_p  + (size_t)B * H;
    float* base  = coef + (size_t)B * H;
    float* mx    = base + (size_t)B * H;
    float* sinv  = mx   + (size_t)B * T;

    gu_precompute<<<dim3(B), dim3(256), 0, stream>>>(dur, vars, c_p, coef, base, H);

    dim3 g2((T + 255) / 256, B);
    gu_stats<<<g2, dim3(256), 0, stream>>>(c_p, coef, base, lengths, mx, sinv, H, T);

    dim3 g3((D + BN - 1) / BN, (T + BM - 1) / BM, B);
    gu_matmul<<<g3, dim3(NTHREADS), 0, stream>>>(enc, lengths, c_p, coef, base,
                                                 mx, sinv, out, H, D, T);
}
